// CRF_2199023255891
// MI455X (gfx1250) — compile-verified, matrix operand order checked by reading, not run on hardware
//
#include <hip/hip_runtime.h>
#include <cstdint>

#define B_ 512
#define T_ 1024
#define L_ 48
#define GROUPS (B_/16)

typedef __attribute__((ext_vector_type(16))) _Float16 v16h;
typedef __attribute__((ext_vector_type(8)))  float    v8f;

#define LOG2E 1.4426950408889634f
#define LN2   0.6931471805599453f

__device__ __forceinline__ float fast_exp2(float x) { return __builtin_amdgcn_exp2f(x); }
__device__ __forceinline__ float fast_log2(float x) { return __builtin_amdgcn_logf(x); }

// One wave advances 16 CRF chains. Transposed formulation: S^T = E^T x exp(a)^T,
// so the constant E^T is the WMMA A operand and the WMMA D output layout feeds the
// next step's B operand directly (no LDS transpose; alpha lives in VGPRs).
// Lane (nlo, half) holds, for batch b0+nlo, states m = mt*16 + half*8 + r (r=0..7).
__global__ __launch_bounds__(32)
void crf_scan_kernel(const float* __restrict__ y_pred,
                     const float* __restrict__ trans,
                     float* __restrict__ logZ)
{
    __shared__ float s_trans[L_ * L_];
    __shared__ float s_tc[L_];

    const int lane = threadIdx.x;
    const int b0   = blockIdx.x * 16;
    const int nlo  = lane & 15;
    const int half = lane >> 4;

    // ---- stage trans, per-column max tc[m] ----
    for (int i = lane; i < L_ * L_; i += 32) s_trans[i] = trans[i];
    __syncthreads();
    for (int c = lane; c < L_; c += 32) {
        float m = s_trans[c];
        for (int l = 1; l < L_; ++l) m = fmaxf(m, s_trans[l * L_ + c]);
        s_tc[c] = m;
    }
    __syncthreads();

    // ---- constant A fragments: E^T, tile (mt,kt): row m = mt*16 + nlo,
    //      element j -> k = kt*32 + j + (j>=8)*8 + half*8 ; E[k][m] = exp(trans[k][m]-tc[m])
    v16h ef[3][2];
    for (int mt = 0; mt < 3; ++mt)
        for (int kt = 0; kt < 2; ++kt) {
            const int m = mt * 16 + nlo;
            for (int j = 0; j < 16; ++j) {
                const int k = kt * 32 + j + ((j >= 8) ? 8 : 0) + half * 8;
                float v = 0.0f;
                if (k < L_) v = fast_exp2((s_trans[k * L_ + m] - s_tc[m]) * LOG2E);
                ef[mt][kt][j] = (_Float16)v;
            }
        }

    // tc (base-2) for this lane's held states
    float tc2[3][8];
    for (int mt = 0; mt < 3; ++mt)
        for (int r = 0; r < 8; ++r)
            tc2[mt][r] = s_tc[mt * 16 + half * 8 + r] * LOG2E;

    // ---- alpha in base-2 units, register resident: al[mt][r] ----
    const float* eptr = y_pred + (size_t)(b0 + nlo) * T_ * L_;
    float al[3][8];
    for (int mt = 0; mt < 3; ++mt)
        for (int q = 0; q < 2; ++q) {
            const float4 v = *(const float4*)(eptr + mt * 16 + half * 8 + q * 4);
            al[mt][q * 4 + 0] = v.x * LOG2E;
            al[mt][q * 4 + 1] = v.y * LOG2E;
            al[mt][q * 4 + 2] = v.z * LOG2E;
            al[mt][q * 4 + 3] = v.w * LOG2E;
        }

    for (int t = 1; t < T_; ++t) {
        // off-chain: base = tc2 + emit*LOG2E (contiguous 8-float runs -> b128 loads)
        float base[3][8];
        for (int mt = 0; mt < 3; ++mt)
            for (int q = 0; q < 2; ++q) {
                const float4 v = *(const float4*)(eptr + (size_t)t * L_ + mt * 16 + half * 8 + q * 4);
                base[mt][q * 4 + 0] = fmaf(v.x, LOG2E, tc2[mt][q * 4 + 0]);
                base[mt][q * 4 + 1] = fmaf(v.y, LOG2E, tc2[mt][q * 4 + 1]);
                base[mt][q * 4 + 2] = fmaf(v.z, LOG2E, tc2[mt][q * 4 + 2]);
                base[mt][q * 4 + 3] = fmaf(v.w, LOG2E, tc2[mt][q * 4 + 3]);
            }
        if (t + 1 < T_)
            __builtin_prefetch(eptr + (size_t)(t + 1) * L_, 0, 3); // WGP-scope prefetch

        // per-batch max: 5-deep register tree over this lane's 24, then cross-half shfl
        float t0[12];
        for (int i = 0; i < 8; ++i) t0[i] = fmaxf(al[0][i], al[1][i]);
        for (int i = 0; i < 4; ++i) t0[8 + i] = fmaxf(al[2][i], al[2][4 + i]);
        float t1[6];
        for (int i = 0; i < 6; ++i) t1[i] = fmaxf(t0[i], t0[6 + i]);
        float t2[3];
        for (int i = 0; i < 3; ++i) t2[i] = fmaxf(t1[i], t1[3 + i]);
        float mx = fmaxf(fmaxf(t2[0], t2[1]), t2[2]);
        mx = fmaxf(mx, __shfl_xor(mx, 16, 32));

        // B fragments: exp2(al - mx), K padded 48->64
        v16h bf0, bf1;
        for (int j = 0; j < 8; ++j) {
            bf0[j]     = (_Float16)fast_exp2(al[0][j] - mx);
            bf0[j + 8] = (_Float16)fast_exp2(al[1][j] - mx);
            bf1[j]     = (_Float16)fast_exp2(al[2][j] - mx);
            bf1[j + 8] = (_Float16)0.0f;
        }

        // S^T tiles: two independent accumulators per M-tile (no WMMA->WMMA RAW)
        const v8f z = {0.f,0.f,0.f,0.f,0.f,0.f,0.f,0.f};
        v8f sa[3], sb[3];
        sa[0] = __builtin_amdgcn_wmma_f32_16x16x32_f16(false, ef[0][0], false, bf0, (short)0, z, false, false);
        sb[0] = __builtin_amdgcn_wmma_f32_16x16x32_f16(false, ef[0][1], false, bf1, (short)0, z, false, false);
        sa[1] = __builtin_amdgcn_wmma_f32_16x16x32_f16(false, ef[1][0], false, bf0, (short)0, z, false, false);
        sb[1] = __builtin_amdgcn_wmma_f32_16x16x32_f16(false, ef[1][1], false, bf1, (short)0, z, false, false);
        sa[2] = __builtin_amdgcn_wmma_f32_16x16x32_f16(false, ef[2][0], false, bf0, (short)0, z, false, false);
        sb[2] = __builtin_amdgcn_wmma_f32_16x16x32_f16(false, ef[2][1], false, bf1, (short)0, z, false, false);

        // alpha' = mx + log2(S) + tc2 + emit*LOG2E  (stays in registers)
        for (int mt = 0; mt < 3; ++mt)
            for (int r = 0; r < 8; ++r)
                al[mt][r] = mx + fast_log2(sa[mt][r] + sb[mt][r]) + base[mt][r];
    }

    // ---- final logsumexp per batch (convert back to nats) ----
    {
        float t0[12];
        for (int i = 0; i < 8; ++i) t0[i] = fmaxf(al[0][i], al[1][i]);
        for (int i = 0; i < 4; ++i) t0[8 + i] = fmaxf(al[2][i], al[2][4 + i]);
        float t1[6];
        for (int i = 0; i < 6; ++i) t1[i] = fmaxf(t0[i], t0[6 + i]);
        float t2[3];
        for (int i = 0; i < 3; ++i) t2[i] = fmaxf(t1[i], t1[3 + i]);
        float mx = fmaxf(fmaxf(t2[0], t2[1]), t2[2]);
        mx = fmaxf(mx, __shfl_xor(mx, 16, 32));

        float s = 0.f;
        for (int mt = 0; mt < 3; ++mt)
            for (int r = 0; r < 8; ++r)
                s += fast_exp2(al[mt][r] - mx);
        s += __shfl_xor(s, 16, 32);
        if (half == 0) logZ[b0 + nlo] = (mx + fast_log2(s)) * LN2;
    }
}

// point score + transition score + final combine (bandwidth-bound)
__global__ __launch_bounds__(256)
void crf_score_kernel(const float* __restrict__ y_true,
                      const float* __restrict__ y_pred,
                      const float* __restrict__ trans,
                      const float* __restrict__ logZ,
                      float* __restrict__ out)
{
    __shared__ float s_trans[L_ * L_];
    __shared__ int   s_lab[T_];
    __shared__ float s_red[8];

    const int tid = threadIdx.x;
    const int b   = blockIdx.x;

    for (int i = tid; i < L_ * L_; i += 256) s_trans[i] = trans[i];

    float pt = 0.f;
    for (int t = tid; t < T_; t += 256) {
        const float4* rowp = (const float4*)(y_true + ((size_t)b * T_ + t) * L_);
        int lab = 0;
        for (int q = 0; q < 12; ++q) {
            const float4 v = rowp[q];
            if (v.x > 0.5f) lab = 4 * q + 0;
            if (v.y > 0.5f) lab = 4 * q + 1;
            if (v.z > 0.5f) lab = 4 * q + 2;
            if (v.w > 0.5f) lab = 4 * q + 3;
        }
        s_lab[t] = lab;
        pt += y_pred[((size_t)b * T_ + t) * L_ + lab];
    }
    __syncthreads();

    float ts = 0.f;
    for (int t = tid + 1; t < T_; t += 256)
        ts += s_trans[s_lab[t - 1] * L_ + s_lab[t]];

    float v = pt + ts;
    for (int off = 16; off > 0; off >>= 1) v += __shfl_xor(v, off, 32);
    if ((tid & 31) == 0) s_red[tid >> 5] = v;
    __syncthreads();
    if (tid == 0) {
        float tot = 0.f;
        for (int w = 0; w < 8; ++w) tot += s_red[w];
        out[b] = logZ[b] - tot;
    }
}

extern "C" void kernel_launch(void* const* d_in, const int* in_sizes, int n_in,
                              void* d_out, int out_size, void* d_ws, size_t ws_size,
                              hipStream_t stream) {
    (void)in_sizes; (void)n_in; (void)out_size; (void)ws_size;
    const float* y_true = (const float*)d_in[0];
    const float* y_pred = (const float*)d_in[1];
    const float* trans  = (const float*)d_in[2];
    float* out  = (float*)d_out;
    float* logZ = (float*)d_ws;   // 512 floats of scratch

    crf_scan_kernel<<<GROUPS, 32, 0, stream>>>(y_pred, trans, logZ);
    crf_score_kernel<<<B_, 256, 0, stream>>>(y_true, y_pred, trans, logZ, out);
}
